// BinaryLinear_35905926594624
// MI455X (gfx1250) — matrix operand order repaired and checked by reading, not compiled, not run on hardware
//
#include <hip/hip_runtime.h>
#include <hip/hip_bf16.h>
#include <stdint.h>

// ---------------------------------------------------------------------------
// BinaryLinear: out = x @ sign(W)^T + bias
//   pass 1: x -> bf16 (RNE), W -> sign(W) as bf16 (+/-1.0), into d_ws
//   pass 2: bf16 WMMA GEMM, f32 accumulate, async-to-LDS staging
// ---------------------------------------------------------------------------

typedef __attribute__((ext_vector_type(16))) __bf16          v16bf;
typedef __attribute__((ext_vector_type(8)))  float           v8f;
typedef __attribute__((ext_vector_type(8)))  unsigned short  us8;

#define NROWS 8192
#define KDIM  4096
#define ODIM  4096

#define BM 256          // x-rows per block
#define BN 128          // out-cols per block
#define BK 32           // K per stage (bf16)
#define LDSS 40         // halfs per LDS row (80 B, padded vs 64 B payload)

#define ASTAGEH (BM * LDSS)     // halfs per A stage
#define BSTAGEH (BN * LDSS)     // halfs per B stage
#define ASTAGEB (ASTAGEH * 2)   // bytes per A stage (20480)
#define BSTAGEB (BSTAGEH * 2)   // bytes per B stage (10240)

union Frag16 { us8 h[2]; v16bf v; };

__device__ __forceinline__ unsigned short f32_to_bf16_rne(float f) {
    union { float f; uint32_t u; } cv; cv.f = f;
    uint32_t u = cv.u;
    return (unsigned short)((u + 0x7FFFu + ((u >> 16) & 1u)) >> 16);
}

// ---- pass 1a: x (f32) -> bf16 ---------------------------------------------
__global__ void cvt_x_kernel(const float* __restrict__ x,
                             unsigned short* __restrict__ xb, int n) {
    int i = (blockIdx.x * blockDim.x + threadIdx.x) * 8;
    if (i >= n) return;
    const float4* p = (const float4*)(x + i);
    float4 a = p[0], b = p[1];
    us8 o;
    o[0] = f32_to_bf16_rne(a.x); o[1] = f32_to_bf16_rne(a.y);
    o[2] = f32_to_bf16_rne(a.z); o[3] = f32_to_bf16_rne(a.w);
    o[4] = f32_to_bf16_rne(b.x); o[5] = f32_to_bf16_rne(b.y);
    o[6] = f32_to_bf16_rne(b.z); o[7] = f32_to_bf16_rne(b.w);
    *(us8*)(xb + i) = o;
}

// ---- pass 1b: W (f32) -> sign(W) in bf16 (+1.0 / -1.0; sign(0)=+1) --------
__global__ void cvt_w_kernel(const float* __restrict__ w,
                             unsigned short* __restrict__ wb, int n) {
    int i = (blockIdx.x * blockDim.x + threadIdx.x) * 8;
    if (i >= n) return;
    const float4* p = (const float4*)(w + i);
    float4 a = p[0], b = p[1];
    us8 o;
    o[0] = (a.x >= 0.0f) ? 0x3F80u : 0xBF80u;
    o[1] = (a.y >= 0.0f) ? 0x3F80u : 0xBF80u;
    o[2] = (a.z >= 0.0f) ? 0x3F80u : 0xBF80u;
    o[3] = (a.w >= 0.0f) ? 0x3F80u : 0xBF80u;
    o[4] = (b.x >= 0.0f) ? 0x3F80u : 0xBF80u;
    o[5] = (b.y >= 0.0f) ? 0x3F80u : 0xBF80u;
    o[6] = (b.z >= 0.0f) ? 0x3F80u : 0xBF80u;
    o[7] = (b.w >= 0.0f) ? 0x3F80u : 0xBF80u;
    *(us8*)(wb + i) = o;
}

// Async staging: 4x16B of A row + 2x16B of B row, direct global->LDS.
// the immediate offset is added to both the LDS and the global address,
// so one LDS-offset VGPR + one 64-bit addr pair serves all chunks.
__device__ __forceinline__ void stage_async(const unsigned short* gA,
                                            const unsigned short* gB,
                                            unsigned aDst, unsigned bDst) {
    uint64_t a64 = (uint64_t)(uintptr_t)gA;
    uint64_t b64 = (uint64_t)(uintptr_t)gB;
    asm volatile(
        "global_load_async_to_lds_b128 %0, %2, off\n\t"
        "global_load_async_to_lds_b128 %0, %2, off offset:16\n\t"
        "global_load_async_to_lds_b128 %0, %2, off offset:32\n\t"
        "global_load_async_to_lds_b128 %0, %2, off offset:48\n\t"
        "global_load_async_to_lds_b128 %1, %3, off\n\t"
        "global_load_async_to_lds_b128 %1, %3, off offset:16"
        :
        : "v"(aDst), "v"(bDst), "v"(a64), "v"(b64)
        : "memory");
}

__device__ __forceinline__ void wait_async_all() {
    asm volatile("s_wait_asynccnt 0x0" ::: "memory");
}

// ---- pass 2: tiled bf16 WMMA GEMM -----------------------------------------
// 256 threads = 8 waves (4 M-slabs x 2 N-slabs); each wave owns 64x64 out
// = 4x4 tiles of 16x16 -> 16 v_wmma_f32_16x16x32_bf16 per K-step.
__global__ __launch_bounds__(256)
void bgemm_wmma_kernel(const unsigned short* __restrict__ Xb,
                       const unsigned short* __restrict__ Wb,
                       const float* __restrict__ bias,
                       float* __restrict__ out) {
    __shared__ alignas(16) unsigned short As[2 * ASTAGEH];
    __shared__ alignas(16) unsigned short Bs[2 * BSTAGEH];

    const int t    = threadIdx.x;
    const int wave = t >> 5;
    const int lane = t & 31;
    const int ln   = lane & 15;   // A row / B,C,D column within a 16-tile
    const int hi   = lane >> 4;   // half-wave: K-block (A,B) / M-block (C,D)
    const int wm   = wave >> 1;   // 0..3 : 64-row slab
    const int wn   = wave & 1;    // 0..1 : 64-col slab

    const int bm = blockIdx.y * BM;
    const int bn = blockIdx.x * BN;

    // Staging map: A -> 1 thread per row (64 B); B -> 2 threads per row (32 B).
    const unsigned short* gA = Xb + (size_t)(bm + t) * KDIM;
    const unsigned short* gB = Wb + (size_t)(bn + (t >> 1)) * KDIM + (t & 1) * 16;

    // Raw LDS byte offsets (low 32 bits of the flat shared address).
    const unsigned aDst = (unsigned)(uintptr_t)&As[0]
                        + (unsigned)t * (LDSS * 2);
    const unsigned bDst = (unsigned)(uintptr_t)&Bs[0]
                        + (unsigned)(t >> 1) * (LDSS * 2) + (t & 1) * 32;

    // Per-lane fragment base pointers (row and hi folded in); in-loop ds_load
    // addresses are base + cur*stage with pure immediate offsets.
    const unsigned short* aF0 = &As[(wm * 64 + ln) * LDSS + hi * 8];
    const unsigned short* bF0 = &Bs[(wn * 64 + ln) * LDSS + hi * 8];

    v8f acc[4][4] = {};

    // preload stage 0
    stage_async(gA, gB, aDst, bDst);
    wait_async_all();
    __syncthreads();

    const unsigned short* gAk = gA + BK;   // next stage to fetch
    const unsigned short* gBk = gB + BK;

    const int STEPS = KDIM / BK;   // 128
    #pragma unroll 1
    for (int ks = 0; ks < STEPS; ++ks) {
        const int cur = ks & 1;
        const int nxt = cur ^ 1;

        // kick off next stage; overlaps with the WMMA burst below
        if (ks + 1 < STEPS) {
            stage_async(gAk, gBk, aDst + nxt * ASTAGEB, bDst + nxt * BSTAGEB);
            gAk += BK;
            gBk += BK;
        }

        const unsigned short* aS = aF0 + cur * ASTAGEH;
        const unsigned short* bS = bF0 + cur * BSTAGEH;

        // A frags (16x32 bf16): elems 0..7 -> K=hi*8+e ; 8..15 -> K=16+hi*8+e-8
        Frag16 af[4];
        #pragma unroll
        for (int mi = 0; mi < 4; ++mi) {
            af[mi].h[0] = *(const us8*)(aS + mi * (16 * LDSS));
            af[mi].h[1] = *(const us8*)(aS + mi * (16 * LDSS) + 16);
        }
        // B frags (32x16 bf16): lane = col N; elems 0..7 -> K=hi*8+e ; 8..15 -> K=16+hi*8+e-8
        Frag16 bfr[4];
        #pragma unroll
        for (int ni = 0; ni < 4; ++ni) {
            bfr[ni].h[0] = *(const us8*)(bS + ni * (16 * LDSS));
            bfr[ni].h[1] = *(const us8*)(bS + ni * (16 * LDSS) + 16);
        }

        #pragma unroll
        for (int mi = 0; mi < 4; ++mi)
            #pragma unroll
            for (int ni = 0; ni < 4; ++ni)
                acc[mi][ni] = __builtin_amdgcn_wmma_f32_16x16x32_bf16(
                    false, af[mi].v, false, bfr[ni].v,
                    (short)0, acc[mi][ni], false, false);

        wait_async_all();   // my stage-nxt writes landed in LDS
        __syncthreads();    // everyone's landed; stage-cur fully consumed
    }

    // Epilogue. C/D layout: VGPR r -> M = r + 8*hi ; N = ln.
    #pragma unroll
    for (int ni = 0; ni < 4; ++ni) {
        const int n = bn + wn * 64 + ni * 16 + ln;
        const float bv = bias[n];
        #pragma unroll
        for (int mi = 0; mi < 4; ++mi) {
            const int mbase = bm + wm * 64 + mi * 16 + 8 * hi;
            #pragma unroll
            for (int r = 0; r < 8; ++r)
                out[(size_t)(mbase + r) * ODIM + n] = acc[mi][ni][r] + bv;
        }
    }
}

// ---- fallback (only if workspace too small): correct but slow -------------
__global__ void bgemm_naive_kernel(const float* __restrict__ x,
                                   const float* __restrict__ w,
                                   const float* __restrict__ bias,
                                   float* __restrict__ out) {
    int n = blockIdx.x * blockDim.x + threadIdx.x;
    int m = blockIdx.y;
    float s = 0.0f;
    const float* xr = x + (size_t)m * KDIM;
    const float* wr = w + (size_t)n * KDIM;
    for (int k = 0; k < KDIM; ++k)
        s += (wr[k] >= 0.0f) ? xr[k] : -xr[k];
    out[(size_t)m * ODIM + n] = s + bias[n];
}

extern "C" void kernel_launch(void* const* d_in, const int* in_sizes, int n_in,
                              void* d_out, int out_size, void* d_ws, size_t ws_size,
                              hipStream_t stream) {
    const float* x    = (const float*)d_in[0];
    const float* w    = (const float*)d_in[1];
    const float* bias = (const float*)d_in[2];
    float* out = (float*)d_out;

    const size_t xb_bytes = (size_t)NROWS * KDIM * 2;  // 64 MB
    const size_t wb_bytes = (size_t)ODIM  * KDIM * 2;  // 32 MB

    if (ws_size >= xb_bytes + wb_bytes) {
        unsigned short* xb = (unsigned short*)d_ws;
        unsigned short* wb = (unsigned short*)((char*)d_ws + xb_bytes);

        const int nx = NROWS * KDIM;
        const int nw = ODIM * KDIM;
        cvt_x_kernel<<<nx / (8 * 256), 256, 0, stream>>>(x, xb, nx);
        cvt_w_kernel<<<nw / (8 * 256), 256, 0, stream>>>(w, wb, nw);

        dim3 grid(ODIM / BN, NROWS / BM);  // (32, 32)
        bgemm_wmma_kernel<<<grid, 256, 0, stream>>>(xb, wb, bias, out);
    } else {
        dim3 grid(ODIM / 256, NROWS);
        bgemm_naive_kernel<<<grid, 256, 0, stream>>>(x, w, bias, out);
    }
}
